// sparseconnect_layer_86638080295021
// MI455X (gfx1250) — compile-verified, matrix operand order checked by reading, not run on hardware
//
#include <hip/hip_runtime.h>

// ---------------- problem constants (from reference) ----------------
#define BATCH   16384
#define IN_DIM  1024
#define UNITS   512
#define K_CONN  32

// ---------------- CDNA5 WMMA vector types ----------------
typedef __attribute__((ext_vector_type(16))) __bf16 v16bf;
typedef __attribute__((ext_vector_type(8)))  __bf16 v8bf;
typedef __attribute__((ext_vector_type(8)))  float  v8f;

// native (gcc-style) vectors for builtin signatures
typedef int            i4v  __attribute__((vector_size(16)));
typedef float          f4v  __attribute__((vector_size(16)));
typedef unsigned short us4v __attribute__((vector_size(8)));

// ---------------- gfx1250 async-to-LDS support (probe via __has_builtin) ----
#if defined(__has_builtin)
#  if __has_builtin(__builtin_amdgcn_global_load_async_to_lds_b128)
#    define HAVE_ASYNC_LDS 1
#  else
#    define HAVE_ASYNC_LDS 0
#  endif
#else
#  define HAVE_ASYNC_LDS 0
#endif

#if HAVE_ASYNC_LDS
#  if __has_builtin(__builtin_amdgcn_s_wait_asynccnt)
#    define WAIT_ASYNC(n) __builtin_amdgcn_s_wait_asynccnt(n)
#  else
#    define WAIT_ASYNC(n) asm volatile("s_wait_asynccnt %0" ::"i"(n) : "memory")
#  endif
#else
#  define WAIT_ASYNC(n)
#endif

#define AS1 __attribute__((address_space(1)))
#define AS3 __attribute__((address_space(3)))

// 16-byte global -> LDS copy; async (ASYNCcnt-tracked) when available.
static __device__ __forceinline__ void copy16_g2l(const unsigned short* gp,
                                                  unsigned short* lp) {
#if HAVE_ASYNC_LDS
    AS1 i4v* g = (AS1 i4v*)(unsigned long long)(uintptr_t)gp;
    AS3 i4v* l = (AS3 i4v*)(unsigned int)(uintptr_t)lp;
    __builtin_amdgcn_global_load_async_to_lds_b128(g, l, 0, 0);
#else
    *(i4v*)lp = *(const i4v*)gp;
#endif
}

// ---------------- bf16 helpers (RNE split) ----------------
static __device__ __forceinline__ unsigned short f2bf(float f) {
    unsigned int u = __float_as_uint(f);
    u += 0x7FFFu + ((u >> 16) & 1u);     // round-to-nearest-even
    return (unsigned short)(u >> 16);
}
static __device__ __forceinline__ float bf2f(unsigned short h) {
    return __uint_as_float(((unsigned int)h) << 16);
}

// Build a 16-bit WMMA fragment for this lane from two contiguous 16B chunks:
// elements 0..7  -> K = half*8 + {0..7},  elements 8..15 -> K = 16 + half*8 + {0..7}
static __device__ __forceinline__ v16bf load_frag(const unsigned short* p) {
    v8bf lo = *(const v8bf*)p;
    v8bf hi = *(const v8bf*)(p + 16);
    return __builtin_shufflevector(lo, hi,
        0, 1, 2, 3, 4, 5, 6, 7, 8, 9, 10, 11, 12, 13, 14, 15);
}

static __device__ __forceinline__ v8f wmma_bf16(v16bf a, v16bf b, v8f c) {
    return __builtin_amdgcn_wmma_f32_16x16x32_bf16(
        false, a, false, b, (short)0, c, false, false);
}

// ---------------- kernel 1: fp32 x -> bf16 hi/lo split ----------------
__global__ __launch_bounds__(256)
void split_x_kernel(const f4v* __restrict__ x4,
                    us4v* __restrict__ xh4,
                    us4v* __restrict__ xl4) {
    int i = blockIdx.x * 256 + threadIdx.x;   // one float4 per thread
    f4v v = __builtin_nontemporal_load(&x4[i]);  // x fp32 read exactly once
    us4v h, l;
    #pragma unroll
    for (int c = 0; c < 4; ++c) {
        unsigned short hh = f2bf(v[c]);
        h[c] = hh;
        l[c] = f2bf(v[c] - bf2f(hh));
    }
    xh4[i] = h;     // re-read by GEMM: keep RT so it stays L2-resident
    xl4[i] = l;
}

// ---------------- kernel 2: top-32 hard mask, masked weights (bf16 hi/lo) ----
__global__ __launch_bounds__(256)
void topk_mask_kernel(const float* __restrict__ D,
                      const float* __restrict__ GN,
                      const float* __restrict__ W,
                      unsigned short* __restrict__ AWh,
                      unsigned short* __restrict__ AWl) {
    __shared__ float pert[IN_DIM];
    __shared__ float amask[IN_DIM];
    __shared__ float redv[256];
    __shared__ int   redi[256];

    const int u = blockIdx.x;
    const int t = threadIdx.x;
    const size_t base = (size_t)u * IN_DIM;

    #pragma unroll
    for (int j = 0; j < 4; ++j) {
        int i = t + j * 256;
        pert[i]  = D[base + i] + GN[base + i];
        amask[i] = 0.0f;
    }
    __syncthreads();

    for (int s = 0; s < K_CONN; ++s) {
        float bv = -1e30f; int bi = 0;
        #pragma unroll
        for (int j = 0; j < 4; ++j) {
            int i = t + j * 256;
            float v = pert[i];
            if (v > bv) { bv = v; bi = i; }
        }
        redv[t] = bv; redi[t] = bi;
        __syncthreads();
        for (int off = 128; off > 0; off >>= 1) {
            if (t < off) {
                float ov = redv[t + off]; int oi = redi[t + off];
                if (ov > redv[t] || (ov == redv[t] && oi < redi[t])) {
                    redv[t] = ov; redi[t] = oi;
                }
            }
            __syncthreads();
        }
        if (t == 0) {
            int sel = redi[0];
            amask[sel] = 1.0f;
            pert[sel]  = -1e30f;
        }
        __syncthreads();
    }

    #pragma unroll
    for (int j = 0; j < 4; ++j) {
        int i = t + j * 256;
        float aw = amask[i] * W[base + i];
        unsigned short h = f2bf(aw);
        AWh[base + i] = h;
        AWl[base + i] = f2bf(aw - bf2f(h));
    }
}

// ---------------- kernel 3: y = x @ AW^T + b, split-bf16 WMMA --------------
// Block: 256 threads = 8 wave32s (4M x 2N). Block tile 128(M) x 64(N).
// Wave tile 32x32 = 2x2 C tiles; per K-step: acc += xh*wh + xh*wl + xl*wh.
// K-step 32, double-buffered LDS staging via async global->LDS B128 copies.
#define ROWPAD 40   // 80B row stride: 20-bank step, conflict-free, 16B aligned

__global__ __launch_bounds__(256)
void gemm_bf16x3_kernel(const unsigned short* __restrict__ xh,
                        const unsigned short* __restrict__ xl,
                        const unsigned short* __restrict__ wh,
                        const unsigned short* __restrict__ wl,
                        const float* __restrict__ bias,
                        float* __restrict__ y) {
    __shared__ __align__(16) unsigned short sA[2][2][128][ROWPAD]; // [buf][h/l][row][col]
    __shared__ __align__(16) unsigned short sB[2][2][64][ROWPAD];

    const int t     = threadIdx.x;
    const int lane  = t & 31;
    const int wave  = t >> 5;
    const int waveM = wave & 3;          // 0..3
    const int waveN = wave >> 2;         // 0..1
    const int mblk  = blockIdx.x * 128;
    const int nblk  = blockIdx.y * 64;
    const int l16   = lane & 15;
    const int kh    = lane >> 4;         // lane half selects K sub-block

    const unsigned short* xsrc[2] = { xh, xl };
    const unsigned short* wsrc[2] = { wh, wl };

    // stage one 128x32 (A: h+l) + 64x32 (B: h+l) K-slice into LDS buffer nb
    // 1536 16B chunks total -> 6 per thread (4 A-chunks + 2 B-chunks)
    auto stage = [&](int nb, int k0) {
        #pragma unroll
        for (int j = 0; j < 4; ++j) {               // A: chunks 0..1023
            int c   = t + 256 * j;
            int hl  = c >> 9;
            int cc  = c & 511;
            int row = cc >> 2;
            int col = (cc & 3) * 8;
            copy16_g2l(xsrc[hl] + (size_t)(mblk + row) * IN_DIM + k0 + col,
                       &sA[nb][hl][row][col]);
        }
        #pragma unroll
        for (int j = 0; j < 2; ++j) {               // B: chunks 0..511
            int c   = t + 256 * j;
            int hl  = c >> 8;
            int cc  = c & 255;
            int row = cc >> 2;
            int col = (cc & 3) * 8;
            copy16_g2l(wsrc[hl] + (size_t)(nblk + row) * IN_DIM + k0 + col,
                       &sB[nb][hl][row][col]);
        }
    };

    v8f acc[2][2];
    #pragma unroll
    for (int mi = 0; mi < 2; ++mi)
        #pragma unroll
        for (int ni = 0; ni < 2; ++ni)
            acc[mi][ni] = (v8f){};

    stage(0, 0);

    const int NT = IN_DIM / 32;   // 32 K-steps
    for (int kt = 0; kt < NT; ++kt) {
        const int cur = kt & 1;
        if (kt + 1 < NT) {
            stage(cur ^ 1, (kt + 1) * 32);
            WAIT_ASYNC(6);        // in-order: the 6 current-buffer copies are done
        } else {
            WAIT_ASYNC(0);
        }
        __syncthreads();          // current buffer visible to all waves

        v16bf af[2][2], bf[2][2]; // [h/l][tile]
        #pragma unroll
        for (int mi = 0; mi < 2; ++mi) {
            int row = waveM * 32 + mi * 16 + l16;
            af[0][mi] = load_frag(&sA[cur][0][row][kh * 8]);
            af[1][mi] = load_frag(&sA[cur][1][row][kh * 8]);
        }
        #pragma unroll
        for (int ni = 0; ni < 2; ++ni) {
            int row = waveN * 32 + ni * 16 + l16;
            bf[0][ni] = load_frag(&sB[cur][0][row][kh * 8]);
            bf[1][ni] = load_frag(&sB[cur][1][row][kh * 8]);
        }

        #pragma unroll
        for (int mi = 0; mi < 2; ++mi) {
            #pragma unroll
            for (int ni = 0; ni < 2; ++ni) {
                acc[mi][ni] = wmma_bf16(af[0][mi], bf[0][ni], acc[mi][ni]); // hi*hi
                acc[mi][ni] = wmma_bf16(af[0][mi], bf[1][ni], acc[mi][ni]); // hi*lo
                acc[mi][ni] = wmma_bf16(af[1][mi], bf[0][ni], acc[mi][ni]); // lo*hi
            }
        }
        __syncthreads();          // all reads of current buffer complete
    }

    // C/D layout: VGPR v, lanes 0-15 -> M=v, N=lane; lanes 16-31 -> M=8+v.
    // y is written once and never re-read: non-temporal stores keep L2 for x.
    #pragma unroll
    for (int mi = 0; mi < 2; ++mi) {
        #pragma unroll
        for (int ni = 0; ni < 2; ++ni) {
            int ncol = nblk + waveN * 32 + ni * 16 + l16;
            float bv = bias[ncol];
            #pragma unroll
            for (int v = 0; v < 8; ++v) {
                int mrow = mblk + waveM * 32 + mi * 16 + v + 8 * kh;
                __builtin_nontemporal_store(acc[mi][ni][v] + bv,
                                            &y[(size_t)mrow * UNITS + ncol]);
            }
        }
    }
}

// ---------------- launcher ----------------
extern "C" void kernel_launch(void* const* d_in, const int* in_sizes, int n_in,
                              void* d_out, int out_size, void* d_ws, size_t ws_size,
                              hipStream_t stream) {
    const float* x  = (const float*)d_in[0];   // [16384,1024]
    const float* W  = (const float*)d_in[1];   // [512,1024]
    const float* b  = (const float*)d_in[2];   // [512]
    const float* D  = (const float*)d_in[3];   // [512,1024]
    const float* GN = (const float*)d_in[4];   // [1,512,1024]
    float* y = (float*)d_out;                  // [16384,512]

    char* ws = (char*)d_ws;
    const size_t XB  = (size_t)BATCH * IN_DIM * sizeof(unsigned short); // 33.5 MB
    const size_t AWB = (size_t)UNITS * IN_DIM * sizeof(unsigned short); // 1 MB
    unsigned short* xh  = (unsigned short*)(ws);
    unsigned short* xl  = (unsigned short*)(ws + XB);
    unsigned short* awh = (unsigned short*)(ws + 2 * XB);
    unsigned short* awl = (unsigned short*)(ws + 2 * XB + AWB);

    split_x_kernel<<<(BATCH * IN_DIM) / (4 * 256), 256, 0, stream>>>(
        (const f4v*)x, (us4v*)xh, (us4v*)xl);

    topk_mask_kernel<<<UNITS, 256, 0, stream>>>(D, GN, W, awh, awl);

    dim3 grid(BATCH / 128, UNITS / 64);
    gemm_bf16x3_kernel<<<grid, 256, 0, stream>>>(xh, xl, awh, awl, b, y);
}
